// CrossAttention_82274393522807
// MI455X (gfx1250) — matrix operand order repaired and checked by reading, not compiled, not run on hardware
//
#include <hip/hip_runtime.h>
#include <hip/hip_bf16.h>

// ---------------------------------------------------------------------------
// CrossAttention, CDNA5 (gfx1250), wave32 + v_wmma_f32_16x16x32_bf16.
//
// Algebraic reduction: q/k rows are scalar*vec+bias, so softmax over keys
// reduces to softmax_j(s_i * y_j), s_i = A_h*x_i + C_h. Exact row max is
// s_i * (s_i>=0 ? ymax_b : ymin_b)  => single-pass softmax, no rescaling.
// Remaining matmuls (value@Wv and P@V) run on WMMA bf16 w/ f32 accumulate.
// ---------------------------------------------------------------------------

typedef __bf16 bf16_t;
typedef __bf16 v16bf __attribute__((ext_vector_type(16)));
typedef __bf16 v8bf  __attribute__((ext_vector_type(8)));
typedef float  v8f   __attribute__((ext_vector_type(8)));
typedef float  v4f   __attribute__((ext_vector_type(4)));

#define NB    4
#define SQ    2048
#define SK    2048
#define NH    16
#define HD    64
#define NHID  1024

#if defined(__has_builtin) && __has_builtin(__builtin_amdgcn_exp2f)
#define EXP2F(x) __builtin_amdgcn_exp2f(x)
#else
#define EXP2F(x) exp2f(x)
#endif

// ---------------------------------------------------------------------------
// K0: per-head scalars A_h = (Wq_h . Wk_h)/8, C_h = (bq_h . Wk_h)/8,
//     plus per-batch min/max of key scalars (for exact softmax max).
// ---------------------------------------------------------------------------
__global__ void __launch_bounds__(256)
k0_prep(const float* __restrict__ key,
        const float* __restrict__ Wq, const float* __restrict__ bq,
        const float* __restrict__ Wk,
        float* __restrict__ scal)  // [0:16)=A, [16:32)=C, [32:36)=ymin, [36:40)=ymax
{
  const int t = threadIdx.x;
  if (t < NH) {
    float a = 0.f, c = 0.f;
    for (int d = 0; d < HD; ++d) {
      const float wk = Wk[t * HD + d];
      a += Wq[t * HD + d] * wk;
      c += bq[t * HD + d] * wk;
    }
    scal[t]      = a * 0.125f;   // 1/sqrt(64)
    scal[16 + t] = c * 0.125f;
  }
  __shared__ float smin[256], smax[256];
  const int b = t >> 6, l64 = t & 63;
  float mn = 3.0e38f, mx = -3.0e38f;
  for (int k = 0; k < SK / 64; ++k) {
    const float y = key[b * SK + l64 + 64 * k];
    mn = fminf(mn, y);
    mx = fmaxf(mx, y);
  }
  smin[t] = mn; smax[t] = mx;
  __syncthreads();
  for (int off = 32; off > 0; off >>= 1) {
    if (l64 < off) {
      smin[t] = fminf(smin[t], smin[t + off]);
      smax[t] = fmaxf(smax[t], smax[t + off]);
    }
    __syncthreads();
  }
  if (l64 == 0) { scal[32 + b] = smin[t]; scal[36 + b] = smax[t]; }
}

// ---------------------------------------------------------------------------
// K0b: Wvt[n][k] = (bf16) Wv[k][n]  (N-major so WMMA B-operand reads are
//      two contiguous 16B runs per lane).
// ---------------------------------------------------------------------------
__global__ void __launch_bounds__(256)
k0_transpose_wv(const float* __restrict__ Wv, bf16_t* __restrict__ Wvt)
{
  const int idx = blockIdx.x * 256 + threadIdx.x;   // 1M elements
  const int k = idx >> 10, n = idx & (NHID - 1);
  Wvt[(size_t)n * NHID + k] = (bf16_t)Wv[(size_t)k * NHID + n];
}

// ---------------------------------------------------------------------------
// K1: V projection GEMM  C[8192,1024] = value @ Wv + bv, output stored
//     TRANSPOSED per head:  Vt[b][h][d][j] (bf16), j contiguous.
//     Wave tile 16x64 (4 x 16x16x32 WMMA fragments), 8 waves/block => 128x64.
// ---------------------------------------------------------------------------
__global__ void __launch_bounds__(256)
k1_vproj(const float* __restrict__ value, const bf16_t* __restrict__ Wvt,
         const float* __restrict__ bv, bf16_t* __restrict__ Vt)
{
  const int w    = threadIdx.x >> 5;
  const int lane = threadIdx.x & 31;
  const int lm   = lane & 15;         // fragment row/col within tile
  const int lh   = lane >> 4;         // K half-select
  const int mblk = blockIdx.x & 63;   // 64 M-blocks of 128 rows
  const int nblk = blockIdx.x >> 6;   // 16 N-blocks of 64 cols
  const int Mw   = mblk * 128 + w * 16;
  const int Nb   = nblk * 64;
  const int koff = lh * 8;            // A-operand K chunk base
  const int kb   = lh * 8;            // B-operand K chunk base

  const float* arow = value + (size_t)(Mw + lm) * NHID;
  v8f acc[4] = {};

  for (int kk = 0; kk < NHID; kk += 32) {
    // A fragment: lane lm holds row, K = {koff..koff+7} U {koff+16..koff+23}
    v16bf a;
    {
      const v4f* p0 = (const v4f*)(arow + kk + koff);
      const v4f f0 = p0[0], f1 = p0[1];
      const v4f* p1 = (const v4f*)(arow + kk + koff + 16);
      const v4f f2 = p1[0], f3 = p1[1];
#pragma unroll
      for (int i = 0; i < 4; ++i) {
        a[i]      = (bf16_t)f0[i];
        a[4 + i]  = (bf16_t)f1[i];
        a[8 + i]  = (bf16_t)f2[i];
        a[12 + i] = (bf16_t)f3[i];
      }
    }
#pragma unroll
    for (int nt = 0; nt < 4; ++nt) {
      // B fragment: lane lm holds column n, K = {kb..kb+7} U {kb+16..kb+23}
      const bf16_t* brow = Wvt + (size_t)(Nb + nt * 16 + lm) * NHID + kk + kb;
      const v8bf* bp = (const v8bf*)brow;
      const v8bf b0 = bp[0], b1 = bp[2];
      v16bf bb;
#pragma unroll
      for (int i = 0; i < 8; ++i) { bb[i] = b0[i]; bb[8 + i] = b1[i]; }
      acc[nt] = __builtin_amdgcn_wmma_f32_16x16x32_bf16(
          false, a, false, bb, (short)0, acc[nt], false, false);
    }
  }

  // Store transposed: row M = Mw + r + 8*lh  =>  b = M/SK, j = M%SK
  const int batch = Mw / SK;
  const int j0    = (Mw % SK) + 8 * lh;   // consecutive j over r
#pragma unroll
  for (int nt = 0; nt < 4; ++nt) {
    const int n = Nb + nt * 16 + lm;
    const int h = n >> 6, d = n & 63;
    const float bias = bv[n];
    v8bf o;
#pragma unroll
    for (int r = 0; r < 8; ++r) o[r] = (bf16_t)(acc[nt][r] + bias);
    bf16_t* dst = Vt + (((size_t)(batch * NH + h) * HD + d) * SK) + j0;
    *(v8bf*)dst = o;
  }
}

// ---------------------------------------------------------------------------
// K2: attention. Block = one (b,h) pair x 128 queries (8 waves x 16 rows).
//     Single pass over 2048 keys in 32-wide steps:
//       P fragment computed in-register via v_exp_f32 (exact max subtracted),
//       4 WMMA accumulate P@V over d in 4x16 chunks, row-sum l on VALU.
// ---------------------------------------------------------------------------
__global__ void __launch_bounds__(256)
k2_attn(const float* __restrict__ query, const float* __restrict__ key,
        const bf16_t* __restrict__ Vt, const float* __restrict__ scal,
        float* __restrict__ out)
{
  __shared__ float y_sh[SK];        // 8 KB: this batch's key scalars
  __shared__ float l_sh[8][16];

  const int w    = threadIdx.x >> 5;
  const int lane = threadIdx.x & 31;
  const int lm   = lane & 15;
  const int lh   = lane >> 4;
  const int qt   = blockIdx.x & 15;          // query tile (128 rows)
  const int h    = (blockIdx.x >> 4) & 15;   // head
  const int b    = blockIdx.x >> 8;          // batch

  for (int k = threadIdx.x; k < SK; k += 256) y_sh[k] = key[b * SK + k];
  __syncthreads();

  const float A    = scal[h];
  const float C    = scal[16 + h];
  const float ymin = scal[32 + b];
  const float ymax = scal[36 + b];

  const int   qrow = qt * 128 + w * 16 + lm;      // this lane's query row
  const float x    = query[b * SQ + qrow];
  const float s    = A * x + C;
  const float m    = s * (s >= 0.f ? ymax : ymin); // exact row max of s*y_j
  const float L2E  = 1.4426950408889634f;
  const float sl   = s * L2E;
  const float ml   = m * L2E;

  const bf16_t* vbase = Vt + (size_t)(b * NH + h) * HD * SK;  // + d*SK + j
  const int koff = lh * 8;
  const int kb   = lh * 8;

  v8f  acc[4] = {};
  float l = 0.f;

  for (int jb = 0; jb < SK; jb += 32) {
    // P fragment (A-operand layout): p = exp(s*y - m)
    v16bf a;
#pragma unroll
    for (int t = 0; t < 8; ++t) {
      const float p0 = EXP2F(sl * y_sh[jb + koff + t]      - ml);
      const float p1 = EXP2F(sl * y_sh[jb + koff + 16 + t] - ml);
      a[t]     = (bf16_t)p0;
      a[8 + t] = (bf16_t)p1;
      l += p0 + p1;
    }
#pragma unroll
    for (int nt = 0; nt < 4; ++nt) {
      const bf16_t* brow = vbase + (size_t)(nt * 16 + lm) * SK + jb + kb;
      const v8bf* bp = (const v8bf*)brow;
      const v8bf b0 = bp[0], b1 = bp[2];
      v16bf bb;
#pragma unroll
      for (int i = 0; i < 8; ++i) { bb[i] = b0[i]; bb[8 + i] = b1[i]; }
      acc[nt] = __builtin_amdgcn_wmma_f32_16x16x32_bf16(
          false, a, false, bb, (short)0, acc[nt], false, false);
    }
    if (jb + 32 < SK)  // pull next V tile (global_prefetch_b8)
      __builtin_prefetch(vbase + (size_t)lm * SK + jb + 32 + kb, 0, 3);
  }

  // Full row-sum: lh=0 covered K%32 in {0..7,16..23}, lh=1 the complement.
  l += __shfl_xor(l, 16, 32);
  if (lh == 0) l_sh[w][lm] = l;
  __syncthreads();

  // C fragment: M = r + 8*lh (query), N = lm (dim chunk). Normalize + store.
#pragma unroll
  for (int nt = 0; nt < 4; ++nt) {
    const int d = nt * 16 + lm;
#pragma unroll
    for (int r = 0; r < 8; ++r) {
      const int mq  = qt * 128 + w * 16 + r + 8 * lh;
      const float inv = 1.0f / l_sh[w][r + 8 * lh];
      out[((size_t)(h * NB + b) * SQ + mq) * HD + d] = acc[nt][r] * inv;
    }
  }
}

// ---------------------------------------------------------------------------
extern "C" void kernel_launch(void* const* d_in, const int* in_sizes, int n_in,
                              void* d_out, int out_size, void* d_ws, size_t ws_size,
                              hipStream_t stream) {
  const float* query = (const float*)d_in[0];
  const float* key   = (const float*)d_in[1];
  const float* value = (const float*)d_in[2];
  const float* Wq    = (const float*)d_in[3];
  const float* bq    = (const float*)d_in[4];
  const float* Wk    = (const float*)d_in[5];
  // d_in[6] (bk) provably cancels in softmax — unused.
  const float* Wv    = (const float*)d_in[7];
  const float* bv    = (const float*)d_in[8];
  float* out = (float*)d_out;

  // Workspace: [0,256)   per-head scalars + batch y min/max (f32)
  //            [256, +2MB)  Wvt bf16 (1024x1024, N-major)
  //            [.., +16MB)  Vt  bf16 (B,H,D,Sk), j contiguous
  char* ws = (char*)d_ws;
  float*  scal = (float*)ws;
  bf16_t* Wvt  = (bf16_t*)(ws + 256);
  bf16_t* Vt   = (bf16_t*)(ws + 256 + (size_t)NHID * NHID * sizeof(bf16_t));
  (void)in_sizes; (void)n_in; (void)out_size; (void)ws_size;

  k0_prep        <<<1,                       256, 0, stream>>>(key, Wq, bq, Wk, scal);
  k0_transpose_wv<<<(NHID * NHID) / 256,     256, 0, stream>>>(Wv, Wvt);
  k1_vproj       <<<64 * 16,                 256, 0, stream>>>(value, Wvt, bv, Vt);
  k2_attn        <<<NB * NH * (SQ / 128),    256, 0, stream>>>(query, key, Vt, scal, out);
}
